// BiModalF_54915451847206
// MI455X (gfx1250) — matrix operand order repaired and checked by reading, not compiled, not run on hardware
//
#include <hip/hip_runtime.h>
#include <hip/hip_bf16.h>

#define B_ 8
#define C_ 128
#define P_ 1024
#define PROT_ 1024
#define ATOM_ 34
#define H_ 128
#define TN_ 8                 // 8 n-tiles of 16 -> N = 128

typedef __attribute__((ext_vector_type(16))) _Float16 v16h;
typedef __attribute__((ext_vector_type(8)))  _Float16 v8h;
typedef __attribute__((ext_vector_type(8)))  float    v8f;
typedef __attribute__((ext_vector_type(4)))  int      v4i;

#define AS1_ __attribute__((address_space(1)))
#define AS3_ __attribute__((address_space(3)))

#if __has_builtin(__builtin_amdgcn_global_load_async_to_lds_b128)
#define ASYNC_LDS 1
#endif

__device__ __forceinline__ void wait_async_stores() {
#if defined(ASYNC_LDS)
#if __has_builtin(__builtin_amdgcn_s_wait_asynccnt)
    __builtin_amdgcn_s_wait_asynccnt(0);
#else
    asm volatile("s_wait_asynccnt 0x0" ::: "memory");
#endif
#endif
}

// ---------------------------------------------------------------- f32 -> f16
__global__ void cvt_f32_f16(const float* __restrict__ in,
                            _Float16* __restrict__ out, int n) {
    int i = blockIdx.x * blockDim.x + threadIdx.x;
    if (i < n) out[i] = (_Float16)in[i];
}

// --------------------------------------------------- WMMA strip GEMM, N = 128
// out[M,128] = act( A[M,K](f16) * W[128,K](f16)^T + bias[128] )
// Block = 128 threads (4 waves); block covers 64 rows, each wave a 16x128
// strip (8 accumulator tiles). Weight slab W[:,kb..kb+31] (8 KB) is staged
// into a double-buffered LDS slab — via async global->LDS copies when the
// toolchain exposes them (ASYNCcnt path) — one K-step ahead of the compute,
// so the copy hides behind the 8-WMMA burst. All 8 B fragments are preloaded
// into VGPRs before the WMMA burst so ds waits overlap matrix issue.
// LDS row stride = 40 halves (80 B = 20 dwords): 20*r mod 64 distinct for
// r = 0..15 -> conflict-free ds_load_b128 fragment reads.
// act: 0 = none, 1 = sigmoid. outh (optional): f16 copy of the result.
__global__ void gemm_strip_wmma(const _Float16* __restrict__ A,   // [M,K]
                                const _Float16* __restrict__ W,   // [128,K]
                                const float* __restrict__ bias,   // [128]
                                float* __restrict__ out,          // [M,128]
                                _Float16* __restrict__ outh,
                                int M, int K, int act) {
    __shared__ _Float16 Wlds[2][128 * 40];       // 2 x 10240 B
    const int tid  = threadIdx.x;
    const int wave = tid >> 5;
    const int lane = tid & 31;
    const int nLo  = lane & 15;
    const int hi   = lane >> 4;                  // K-half select
    const int m0   = blockIdx.x * 64 + wave * 16;

    union { v8f v; float f[8]; } acc[TN_];
#pragma unroll
    for (int nt = 0; nt < TN_; ++nt) {
        const float bv = bias[nt * 16 + nLo];
#pragma unroll
        for (int r = 0; r < 8; ++r) acc[nt].f[r] = bv;
    }

    const _Float16* __restrict__ Arow = A + (size_t)(m0 + nLo) * K;
    const _Float16* __restrict__ Wrow = W + (size_t)tid * K;   // row `tid` of W

    // stage W[:, kb..kb+31] into slab `buf`: thread t copies its 64-byte row slice
    auto stage = [&](int buf, int kb) {
        const _Float16* src = Wrow + kb;
        _Float16* dst = &Wlds[buf][tid * 40];
#if defined(ASYNC_LDS)
        AS1_ v4i* g = (AS1_ v4i*)src;            // global source (addrspace 1)
        AS3_ v4i* l = (AS3_ v4i*)dst;            // LDS destination (addrspace 3)
        __builtin_amdgcn_global_load_async_to_lds_b128(g,     l,     0, 0);
        __builtin_amdgcn_global_load_async_to_lds_b128(g + 1, l + 1, 0, 0);
        __builtin_amdgcn_global_load_async_to_lds_b128(g + 2, l + 2, 0, 0);
        __builtin_amdgcn_global_load_async_to_lds_b128(g + 3, l + 3, 0, 0);
#else
        const v8h* s = (const v8h*)src;
        v8h* d = (v8h*)dst;
        d[0] = s[0]; d[1] = s[1]; d[2] = s[2]; d[3] = s[3];
#endif
    };

    stage(0, 0);
    int buf = 0;
    for (int kb = 0; kb < K; kb += 32) {
        wait_async_stores();                     // my slab copies landed in LDS
        __syncthreads();                         // everyone's landed
        if (kb + 32 < K) stage(buf ^ 1, kb + 32);   // prefetch next slab (uniform)

        // A 16x32 fragment: lanes 0-15 K{0..7,16..23}, lanes 16-31 K{8..15,24..31}
        union { v16h v; v8h h[2]; } af;
        af.h[0] = *(const v8h*)(Arow + kb + hi * 8);
        af.h[1] = *(const v8h*)(Arow + kb + 16 + hi * 8);
        if (kb + 32 < K) __builtin_prefetch(Arow + kb + 32, 0, 1);

        // preload all 8 B fragments, then burst 8 WMMAs
        union { v16h v; v8h h[2]; } bf[TN_];
#pragma unroll
        for (int nt = 0; nt < TN_; ++nt) {
            const v8h* bsrc =
                (const v8h*)(&Wlds[buf][(nt * 16 + nLo) * 40 + hi * 8]);
            bf[nt].h[0] = bsrc[0];
            bf[nt].h[1] = bsrc[2];
        }
#pragma unroll
        for (int nt = 0; nt < TN_; ++nt) {
            acc[nt].v = __builtin_amdgcn_wmma_f32_16x16x32_f16(
                /*neg_a=*/false, af.v, /*neg_b=*/false, bf[nt].v,
                /*c_mod=*/(short)0, acc[nt].v, /*reuse_a=*/false, /*reuse_b=*/false);
        }
        buf ^= 1;
    }

#pragma unroll
    for (int nt = 0; nt < TN_; ++nt) {
#pragma unroll
        for (int r = 0; r < 8; ++r) {
            float v = acc[nt].f[r];
            if (act == 1) v = 1.0f / (1.0f + __expf(-v));
            size_t idx = (size_t)(m0 + hi * 8 + r) * 128 + nt * 16 + nLo;
            out[idx] = v;
            if (outh) outh[idx] = (_Float16)v;
        }
    }
}

// ---------------------------------------------- cp = compound @ Wc.T + bc (K=34)
__global__ void compound_proj(const float* __restrict__ comp,  // [B*C,34]
                              const float* __restrict__ Wc,    // [H,34]
                              const float* __restrict__ bc,    // [H]
                              float* __restrict__ cp,          // [B*C,H]
                              _Float16* __restrict__ cph) {
    int t = blockIdx.x * blockDim.x + threadIdx.x;
    if (t >= B_ * C_ * H_) return;
    int h   = t & (H_ - 1);
    int row = t >> 7;
    const float* x = comp + (size_t)row * ATOM_;
    const float* w = Wc + (size_t)h * ATOM_;
    float a = bc[h];
#pragma unroll
    for (int k = 0; k < ATOM_; ++k) a += x[k] * w[k];
    cp[t]  = a;
    cph[t] = (_Float16)a;
}

// -------------------- Sc[b,c,h] = mean_p relu(ca[b,c,h] + pa[b,p,h])  (f16 out)
__global__ void pair_sum_c(const float* __restrict__ ca,   // [B*C,H]
                           const float* __restrict__ pa,   // [B*P,H]
                           _Float16* __restrict__ Sc) {    // [B*C,H]
    int t = blockIdx.x * blockDim.x + threadIdx.x;
    if (t >= B_ * C_ * H_) return;
    int h  = t & (H_ - 1);
    int bc = t >> 7;
    int b  = bc >> 7;                  // C_ == 128
    const float cav = ca[t];
    const float* pab = pa + ((size_t)b * P_) * H_ + h;
    float acc = 0.0f;
    for (int p = 0; p < P_; ++p) {
        float v = cav + pab[(size_t)p * H_];
        acc += v > 0.0f ? v : 0.0f;
    }
    Sc[t] = (_Float16)(acc * (1.0f / (float)P_));
}

// -------------------- Sp[b,p,h] = mean_c relu(ca[b,c,h] + pa[b,p,h])  (f16 out)
__global__ void pair_sum_p(const float* __restrict__ ca,
                           const float* __restrict__ pa,
                           _Float16* __restrict__ Sp) {    // [B*P,H]
    int t = blockIdx.x * blockDim.x + threadIdx.x;
    if (t >= B_ * P_ * H_) return;
    int h  = t & (H_ - 1);
    int bp = t >> 7;
    int b  = bp >> 10;                 // P_ == 1024
    const float pav = pa[t];
    const float* cab = ca + ((size_t)b * C_) * H_ + h;
    float acc = 0.0f;
    for (int c = 0; c < C_; ++c) {
        float v = pav + cab[(size_t)c * H_];
        acc += v > 0.0f ? v : 0.0f;
    }
    Sp[t] = (_Float16)(acc * (1.0f / (float)C_));
}

// ---------------- pool[b,h] = max_i src[b,i,h] * (0.5 + w[b,i,h]); L = i-extent
__global__ void pool_max(const float* __restrict__ src,  // [B*L,H]
                         const float* __restrict__ w,    // [B*L,H]
                         float* __restrict__ pair,       // [B,2H] dest slice
                         int L, int off) {
    int t = blockIdx.x * blockDim.x + threadIdx.x;
    if (t >= B_ * H_) return;
    int h = t & (H_ - 1);
    int b = t >> 7;
    size_t base = ((size_t)b * L) * H_ + h;
    float m = -3.4e38f;
    for (int i = 0; i < L; ++i) {
        size_t idx = base + (size_t)i * H_;
        float v = src[idx] * (0.5f + w[idx]);
        m = v > m ? v : m;
    }
    pair[(size_t)b * (2 * H_) + off + h] = m;
}

// ----------------------------- out[b,o] = act( in[b,:K] . W[o,:K] + bias[o] )
// mode: 0 = linear, 1 = leaky_relu(0.01)
__global__ void fc_layer(const float* __restrict__ in,   // [B,K]
                         const float* __restrict__ W,    // [N,K]
                         const float* __restrict__ bias, // [N]
                         float* __restrict__ out,        // [B,N]
                         int K, int N, int mode) {
    int t = blockIdx.x * blockDim.x + threadIdx.x;
    if (t >= B_ * N) return;
    int o = t % N;
    int b = t / N;
    const float* x = in + (size_t)b * K;
    const float* w = W + (size_t)o * K;
    float a = bias[o];
    for (int k = 0; k < K; ++k) a += x[k] * w[k];
    if (mode == 1 && a < 0.0f) a *= 0.01f;
    out[t] = a;
}

extern "C" void kernel_launch(void* const* d_in, const int* in_sizes, int n_in,
                              void* d_out, int out_size, void* d_ws, size_t ws_size,
                              hipStream_t stream) {
    const float* protein = (const float*)d_in[0];   // [8,1024,1024]
    const float* compound= (const float*)d_in[1];   // [8,128,34]
    const float* Wp = (const float*)d_in[2];  const float* bp = (const float*)d_in[3];
    const float* Wc = (const float*)d_in[4];  const float* bc = (const float*)d_in[5];
    const float* Wpa= (const float*)d_in[6];  const float* bpa= (const float*)d_in[7];
    const float* Wca= (const float*)d_in[8];  const float* bca= (const float*)d_in[9];
    const float* Wi = (const float*)d_in[10]; const float* bi = (const float*)d_in[11];
    const float* W1 = (const float*)d_in[12]; const float* b1 = (const float*)d_in[13];
    const float* W2 = (const float*)d_in[14]; const float* b2 = (const float*)d_in[15];
    const float* Wo = (const float*)d_in[16]; const float* bo = (const float*)d_in[17];
    float* outp = (float*)d_out;                     // [8,2]

    char* cur = (char*)d_ws;
    auto alloc = [&](size_t bytes) -> void* {
        void* r = (void*)cur;
        cur += (bytes + 255) & ~(size_t)255;
        return r;
    };
    const int MP = B_ * P_;     // 8192
    const int MC = B_ * C_;     // 1024

    _Float16* prot_h = (_Float16*)alloc((size_t)MP * PROT_ * 2);
    _Float16* Wp_h   = (_Float16*)alloc((size_t)H_ * PROT_ * 2);
    _Float16* Wpa_h  = (_Float16*)alloc((size_t)H_ * H_ * 2);
    _Float16* Wca_h  = (_Float16*)alloc((size_t)H_ * H_ * 2);
    _Float16* Wi_h   = (_Float16*)alloc((size_t)H_ * H_ * 2);
    float*    pp     = (float*)   alloc((size_t)MP * H_ * 4);
    _Float16* pp_h   = (_Float16*)alloc((size_t)MP * H_ * 2);
    float*    pa     = (float*)   alloc((size_t)MP * H_ * 4);
    float*    cp     = (float*)   alloc((size_t)MC * H_ * 4);
    _Float16* cp_h   = (_Float16*)alloc((size_t)MC * H_ * 2);
    float*    ca     = (float*)   alloc((size_t)MC * H_ * 4);
    _Float16* Sc_h   = (_Float16*)alloc((size_t)MC * H_ * 2);
    _Float16* Sp_h   = (_Float16*)alloc((size_t)MP * H_ * 2);
    float*    cw     = (float*)   alloc((size_t)MC * H_ * 4);
    float*    pw     = (float*)   alloc((size_t)MP * H_ * 4);
    float*    pair   = (float*)   alloc((size_t)B_ * 2 * H_ * 4);
    float*    x1     = (float*)   alloc((size_t)B_ * 1024 * 4);
    float*    x2     = (float*)   alloc((size_t)B_ * 512 * 4);

    // 1) f32 -> f16 conversions
    int nprot = MP * PROT_;
    cvt_f32_f16<<<(nprot + 255) / 256, 256, 0, stream>>>(protein, prot_h, nprot);
    cvt_f32_f16<<<(H_ * PROT_ + 255) / 256, 256, 0, stream>>>(Wp, Wp_h, H_ * PROT_);
    cvt_f32_f16<<<(H_ * H_ + 255) / 256, 256, 0, stream>>>(Wpa, Wpa_h, H_ * H_);
    cvt_f32_f16<<<(H_ * H_ + 255) / 256, 256, 0, stream>>>(Wca, Wca_h, H_ * H_);
    cvt_f32_f16<<<(H_ * H_ + 255) / 256, 256, 0, stream>>>(Wi, Wi_h, H_ * H_);

    // 2) pp = protein @ Wp.T + bp          [8192 x 1024] x [128 x 1024]^T
    gemm_strip_wmma<<<MP / 64, 128, 0, stream>>>(prot_h, Wp_h, bp, pp, pp_h,
                                                 MP, PROT_, 0);
    // 3) pa = pp @ Wpa.T + bpa
    gemm_strip_wmma<<<MP / 64, 128, 0, stream>>>(pp_h, Wpa_h, bpa, pa, nullptr,
                                                 MP, H_, 0);
    // 4) cp = compound @ Wc.T + bc  (K=34, scalar)
    compound_proj<<<(MC * H_ + 255) / 256, 256, 0, stream>>>(compound, Wc, bc, cp, cp_h);
    // 5) ca = cp @ Wca.T + bca
    gemm_strip_wmma<<<MC / 64, 128, 0, stream>>>(cp_h, Wca_h, bca, ca, nullptr,
                                                 MC, H_, 0);
    // 6) pair means of relu(ca + pa)  (hoisted out of the Wi matmul)
    pair_sum_c<<<(MC * H_ + 255) / 256, 256, 0, stream>>>(ca, pa, Sc_h);
    pair_sum_p<<<(MP * H_ + 255) / 256, 256, 0, stream>>>(ca, pa, Sp_h);
    // 7) weights = sigmoid(mean @ Wi.T + bi)   (fused sigmoid epilogue)
    gemm_strip_wmma<<<MC / 64, 128, 0, stream>>>(Sc_h, Wi_h, bi, cw, nullptr,
                                                 MC, H_, 1);
    gemm_strip_wmma<<<MP / 64, 128, 0, stream>>>(Sp_h, Wi_h, bi, pw, nullptr,
                                                 MP, H_, 1);
    // 8) max-pool of src*(0.5+w), written directly into the concat buffer
    pool_max<<<(B_ * H_ + 255) / 256, 256, 0, stream>>>(cp, cw, pair, C_, 0);
    pool_max<<<(B_ * H_ + 255) / 256, 256, 0, stream>>>(pp, pw, pair, P_, H_);
    // 9) MLP head (M=8, scalar)
    fc_layer<<<(B_ * 1024 + 255) / 256, 256, 0, stream>>>(pair, W1, b1, x1, 2 * H_, 1024, 1);
    fc_layer<<<(B_ * 512 + 255) / 256, 256, 0, stream>>>(x1, W2, b2, x2, 1024, 512, 1);
    fc_layer<<<1, 64, 0, stream>>>(x2, Wo, bo, outp, 512, 2, 0);
}
